// SparseMoE_66829691126410
// MI455X (gfx1250) — matrix operand order repaired and checked by reading, not compiled, not run on hardware
//
#include <hip/hip_runtime.h>
#include <math.h>

#define TOKENS 16384
#define DIM    512
#define HID    2048
#define OUTD   512
#define NEXP   8
#define CAP    2048

typedef __attribute__((ext_vector_type(16))) __bf16          v16bf;
typedef __attribute__((ext_vector_type(8)))  float           v8f;
typedef __attribute__((ext_vector_type(8)))  unsigned short  u16x8;
typedef __attribute__((ext_vector_type(16))) unsigned short  u16x16;
typedef __attribute__((ext_vector_type(4)))  unsigned int    u32x4;
typedef __attribute__((ext_vector_type(8)))  int             i32x8;
typedef __attribute__((ext_vector_type(4)))  int             i32x4;

#if __has_builtin(__builtin_amdgcn_tensor_load_to_lds) && __has_builtin(__builtin_amdgcn_s_wait_tensorcnt)
#define HAVE_TDM 1
#if __has_include(<hip/amd_detail/amd_gfx1250_TDM.h>)
#define TDM_6ARG 1
#else
#define TDM_6ARG 0
#endif
#else
#define HAVE_TDM 0
#endif

__device__ __forceinline__ unsigned short f2bf(float f) {
  unsigned u = __builtin_bit_cast(unsigned, f);
  u += 0x7fffu + ((u >> 16) & 1u);               // round-to-nearest-even
  return (unsigned short)(u >> 16);
}

// Build a 16-element bf16 fragment from two 16-byte LDS chunks.
__device__ __forceinline__ v16bf ld_frag16(const unsigned short* p0, const unsigned short* p1) {
  u16x8 lo = *(const u16x8*)p0;
  u16x8 hi = *(const u16x8*)p1;
  u16x16 c;
#pragma unroll
  for (int i = 0; i < 8; ++i) { c[i] = lo[i]; c[i + 8] = hi[i]; }
  return __builtin_bit_cast(v16bf, c);
}

#if HAVE_TDM
// Issue a TDM 2D tile load (bf16 elements, data_size code 1) global -> LDS.
// pad_ic: pad_interval code (0=2 dw .. 7=256 dw), pad_ac: pad_amount code (n+1 dw).
__device__ __forceinline__ void tdm_load_2d(unsigned lds_byte_off, const void* gaddr,
                                            unsigned tile_d0, unsigned tile_d1,
                                            unsigned stride_d0,
                                            unsigned tensor_d0, unsigned tensor_d1,
                                            unsigned pad_ic, unsigned pad_ac)
{
  size_t ga = (size_t)gaddr;
  u32x4 g0;
  g0[0] = 1u;                                              // count=1, user-mode load
  g0[1] = lds_byte_off;                                    // D#.lds_addr
  g0[2] = (unsigned)(ga & 0xffffffffu);                    // global_addr[31:0]
  g0[3] = (unsigned)((ga >> 32) & 0x01ffffffu) | (2u << 30); // global_addr[56:32] | type=2
  i32x8 g1;
  g1[0] = (int)((1u << 16) | (1u << 20) | (pad_ic << 22) | (pad_ac << 25)); // data_size=2B, pad_enable
  g1[1] = (int)((tensor_d0 & 0xffffu) << 16);              // tensor_dim0[15:0] @ [79:64... ] low part
  g1[2] = (int)(((tensor_d0 >> 16) & 0xffffu) | ((tensor_d1 & 0xffffu) << 16));
  g1[3] = (int)(((tensor_d1 >> 16) & 0xffffu) | ((tile_d0 & 0xffffu) << 16));
  g1[4] = (int)(tile_d1 & 0xffffu);                        // tile_dim1, tile_dim2=0
  g1[5] = (int)stride_d0;                                  // tensor_dim0_stride[31:0]
  g1[6] = 0;                                               // stride hi / dim1_stride lo
  g1[7] = 0;
  i32x4 zz = {0, 0, 0, 0};
#if TDM_6ARG
  i32x8 z8 = {0, 0, 0, 0, 0, 0, 0, 0};
  __builtin_amdgcn_tensor_load_to_lds(g0, g1, zz, zz, z8, 0);
#else
  __builtin_amdgcn_tensor_load_to_lds(g0, g1, zz, zz, 0);
#endif
}
#endif

// ---------------------------------------------------------------------------
// Gating: logits -> softmax -> probs out, top-2 indices + normalized weights.
// ---------------------------------------------------------------------------
__global__ __launch_bounds__(256) void gate_kernel(
    const float* __restrict__ x, const float* __restrict__ gw, const float* __restrict__ gb,
    float* __restrict__ probs, int* __restrict__ i1, int* __restrict__ i2,
    float* __restrict__ wt1, float* __restrict__ wt2)
{
  __shared__ float gws[DIM * NEXP];
  __shared__ float gbs[NEXP];
  int tid = threadIdx.x;
  for (int i = tid; i < DIM * NEXP; i += 256) gws[i] = gw[i];
  if (tid < NEXP) gbs[tid] = gb[tid];
  __syncthreads();

  int t = blockIdx.x * 256 + tid;
  float acc[NEXP];
#pragma unroll
  for (int e = 0; e < NEXP; ++e) acc[e] = gbs[e];
  const float4* xr = (const float4*)(x + (size_t)t * DIM);
  for (int q = 0; q < DIM / 4; ++q) {
    float4 xv = xr[q];
    const float* g0 = &gws[(q * 4) * NEXP];
#pragma unroll
    for (int e = 0; e < NEXP; ++e)
      acc[e] += xv.x * g0[e] + xv.y * g0[NEXP + e] + xv.z * g0[2 * NEXP + e] + xv.w * g0[3 * NEXP + e];
  }
  float mx = acc[0];
#pragma unroll
  for (int e = 1; e < NEXP; ++e) mx = fmaxf(mx, acc[e]);
  float s = 0.f;
#pragma unroll
  for (int e = 0; e < NEXP; ++e) { acc[e] = expf(acc[e] - mx); s += acc[e]; }
  float inv = 1.f / s;
#pragma unroll
  for (int e = 0; e < NEXP; ++e) { acc[e] *= inv; probs[(size_t)t * NEXP + e] = acc[e]; }

  int a1 = 0; float p1 = acc[0];
#pragma unroll
  for (int e = 1; e < NEXP; ++e) if (acc[e] > p1) { p1 = acc[e]; a1 = e; }
  int a2 = -1; float p2 = -1e30f;
#pragma unroll
  for (int e = 0; e < NEXP; ++e) if (e != a1 && acc[e] > p2) { p2 = acc[e]; a2 = e; }
  float inv2 = 1.f / (p1 + p2);
  i1[t] = a1; i2[t] = a2; wt1[t] = p1 * inv2; wt2[t] = p2 * inv2;
}

// ---------------------------------------------------------------------------
// Per-expert capacity dispatch preserving token order (wave32 ballot scan).
// ---------------------------------------------------------------------------
__global__ void dispatch_kernel(const int* __restrict__ i1, const int* __restrict__ i2,
                                const float* __restrict__ wt1, const float* __restrict__ wt2,
                                int* __restrict__ idx, float* __restrict__ wgt)
{
  int e = blockIdx.x;
  int lane = threadIdx.x;            // 32 threads = 1 wave
  for (int p = lane; p < CAP; p += 32) { idx[e * CAP + p] = 0; wgt[e * CAP + p] = 0.f; }
  int base = 0;
  for (int t0 = 0; t0 < TOKENS; t0 += 32) {
    int t = t0 + lane;
    int a = i1[t], b = i2[t];
    bool sel = (a == e) || (b == e);
    unsigned mask = (unsigned)__ballot(sel);
    int prefix = __popc(mask & ((1u << lane) - 1u));
    int pos = base + prefix;
    if (sel && pos < CAP) {
      idx[e * CAP + pos] = t;
      wgt[e * CAP + pos] = (a == e) ? wt1[t] : wt2[t];
    }
    base += __popc(mask);
    if (base >= CAP) break;          // remaining tokens are all dropped (cumsum > capacity)
  }
}

// ---------------------------------------------------------------------------
// One-time f32 -> bf16 conversions (x elementwise; W transposed via LDS tile).
// ---------------------------------------------------------------------------
__global__ void cvt_bf16_kernel(const float* __restrict__ src, unsigned short* __restrict__ dst, int n) {
  int i = blockIdx.x * blockDim.x + threadIdx.x;
  if (i < n) dst[i] = f2bf(src[i]);
}

__global__ void transpose_bf16_kernel(const float* __restrict__ src, unsigned short* __restrict__ dst,
                                      int R, int C)
{
  // src: [b][R][C] f32 ; dst: [b][C][R] bf16
  __shared__ unsigned short tile[32][33];
  int b = blockIdx.z;
  int r0 = blockIdx.y * 32, c0 = blockIdx.x * 32;
  const float* s = src + (size_t)b * R * C;
  unsigned short* d = dst + (size_t)b * R * C;
  int tx = threadIdx.x;
  for (int j = threadIdx.y; j < 32; j += 8)
    tile[j][tx] = f2bf(s[(size_t)(r0 + j) * C + c0 + tx]);
  __syncthreads();
  for (int j = threadIdx.y; j < 32; j += 8)
    d[(size_t)(c0 + j) * R + r0 + tx] = tile[tx][j];
}

// ---------------------------------------------------------------------------
// Fused expert FFN: out[idx] += ( gelu(X@W1+b1) @ W2 + b2 ) * w, via bf16 WMMA.
// Block = 256 threads (8 waves), 64 tokens x full O=512, chunked over H by 64.
// W chunks staged by the Tensor Data Mover, double-pumped against the WMMAs.
// ---------------------------------------------------------------------------
__global__ __launch_bounds__(256) void moe_ffn_kernel(
    const unsigned short* __restrict__ xbf,   // [T][D]   bf16
    const unsigned short* __restrict__ w1bf,  // [E][H][D] bf16 (transposed)
    const float* __restrict__ b1,             // [E][H]
    const unsigned short* __restrict__ w2bf,  // [E][O][H] bf16 (transposed)
    const float* __restrict__ b2,             // [E][O]
    const int* __restrict__ idx,              // [E][CAP]
    const float* __restrict__ wgt,            // [E][CAP]
    float* __restrict__ out)                  // [T][O]
{
  constexpr int MT = 64, HC = 64;
  constexpr int XP = DIM + 8;   // padded bf16 stride (bank-conflict avoidance)
  constexpr int WP = HC + 8;

  __shared__ __align__(16) unsigned short Xs[MT * XP];     // gathered tokens [row][k]
  __shared__ __align__(16) unsigned short W1s[HC * XP];    // [h'][d]   (B layout GEMM1)
  __shared__ __align__(16) unsigned short W2s[OUTD * WP];  // [o][h']   (B layout GEMM2)
  __shared__ __align__(16) unsigned short Hc[MT * WP];     // gelu intermediate [m][h']
  __shared__ int   idx_s[MT];
  __shared__ float wgt_s[MT];

  const int e    = blockIdx.y;
  const int m0   = blockIdx.x * MT;
  const int tid  = threadIdx.x;
  const int lane = tid & 31;
  const int wv   = tid >> 5;
  const int ln16 = lane & 15;
  const int half = lane >> 4;

  if (tid < MT) { idx_s[tid] = idx[e * CAP + m0 + tid]; wgt_s[tid] = wgt[e * CAP + m0 + tid]; }
  __syncthreads();

  const unsigned short* w1e = w1bf + (size_t)e * HID * DIM;
  const unsigned short* w2e = w2bf + (size_t)e * OUTD * HID;

#if HAVE_TDM
  const unsigned w1s_off = (unsigned)(size_t)&W1s[0];   // generic LDS ptr low bits = LDS offset
  const unsigned w2s_off = (unsigned)(size_t)&W2s[0];
  if (wv == 0) {
    // chunk 0: W1 tile 64 rows x 512 elems (contiguous), pad 4 dw per 256 dw (XP=520).
    tdm_load_2d(w1s_off, w1e, DIM, HC, DIM, DIM, HID, 7u, 3u);
    // chunk 0: W2 tile 512 rows x 64 elems (stride HID), pad 4 dw per 32 dw (WP=72).
    tdm_load_2d(w2s_off, w2e, HC, OUTD, HID, HID, OUTD, 4u, 3u);
  }
#endif

  // Gather X rows (bf16) into LDS, 16B vector copies.
  for (int i = tid; i < MT * (DIM / 8); i += 256) {
    int r  = i >> 6;
    int c8 = (i & 63) * 8;
    *(u16x8*)&Xs[r * XP + c8] = *(const u16x8*)&xbf[(size_t)idx_s[r] * DIM + c8];
  }

#if HAVE_TDM
  if (wv == 0) __builtin_amdgcn_s_wait_tensorcnt((short)1);  // W1(0) landed
  __syncthreads();
#endif

  // Wave tiling: GEMM1 -> 64x64 (2 tiles/wave); GEMM2 -> 64x512 (16 tiles/wave).
  const int tm  = wv & 3;
  const int tnb = (wv >> 2) * 2;
  const int cg  = (wv >> 2) * 256;

  v8f zero = {0.f, 0.f, 0.f, 0.f, 0.f, 0.f, 0.f, 0.f};
  v8f acc[16];
#pragma unroll
  for (int i = 0; i < 16; ++i) acc[i] = zero;

  for (int hc = 0; hc < HID; hc += HC) {
#if !HAVE_TDM
    // Fallback: cooperative vector-copy staging of W chunks.
    for (int i = tid; i < HC * (DIM / 8); i += 256) {
      int h  = i >> 6;
      int d8 = (i & 63) * 8;
      *(u16x8*)&W1s[h * XP + d8] = *(const u16x8*)&w1e[(size_t)(hc + h) * DIM + d8];
    }
    for (int i = tid; i < OUTD * (HC / 8); i += 256) {
      int o  = i >> 3;
      int h8 = (i & 7) * 8;
      *(u16x8*)&W2s[o * WP + h8] = *(const u16x8*)&w2e[(size_t)o * HID + hc + h8];
    }
    if (hc + HC < HID) {
      __builtin_prefetch(&w1e[(size_t)(hc + HC) * DIM + tid * 128], 0, 1);
      __builtin_prefetch(&w2e[(size_t)tid * HID + hc + HC], 0, 1);
      __builtin_prefetch(&w2e[(size_t)(tid + 256) * HID + hc + HC], 0, 1);
    }
    __syncthreads();
#endif

    // GEMM1: H-chunk = X(64x512) @ W1c(512x64), K loop of 16 WMMA steps.
    v8f h0 = zero, h1 = zero;
#pragma unroll
    for (int ks = 0; ks < DIM / 32; ++ks) {
      int kb = ks * 32;
      const unsigned short* ap = &Xs[(tm * 16 + ln16) * XP + kb + half * 16];
      v16bf a = ld_frag16(ap, ap + 8);
      int bk = kb + half * 16;
      const unsigned short* bp0 = &W1s[(tnb * 16 + ln16) * XP + bk];
      const unsigned short* bp1 = &W1s[((tnb + 1) * 16 + ln16) * XP + bk];
      v16bf b0  = ld_frag16(bp0, bp0 + 8);
      v16bf b1v = ld_frag16(bp1, bp1 + 8);
      h0 = __builtin_amdgcn_wmma_f32_16x16x32_bf16(false, a, false, b0,  (short)0, h0, false, false);
      h1 = __builtin_amdgcn_wmma_f32_16x16x32_bf16(false, a, false, b1v, (short)0, h1, false, false);
    }
    // bias + exact GELU + store bf16 intermediate to LDS.
    float bias0 = b1[e * HID + hc + tnb * 16 + ln16];
    float bias1 = b1[e * HID + hc + (tnb + 1) * 16 + ln16];
#pragma unroll
    for (int v = 0; v < 8; ++v) {
      int m = tm * 16 + half * 8 + v;
      float g0 = h0[v] + bias0;
      g0 = 0.5f * g0 * (1.f + erff(g0 * 0.70710678118654752f));
      Hc[m * WP + tnb * 16 + ln16] = f2bf(g0);
      float g1 = h1[v] + bias1;
      g1 = 0.5f * g1 * (1.f + erff(g1 * 0.70710678118654752f));
      Hc[m * WP + (tnb + 1) * 16 + ln16] = f2bf(g1);
    }

#if HAVE_TDM
    if (wv == 0) __builtin_amdgcn_s_wait_tensorcnt((short)0);  // W2(hc) landed
    __syncthreads();                                           // Hc ready, W1s free
    if (wv == 0 && hc + HC < HID)
      tdm_load_2d(w1s_off, w1e + (size_t)(hc + HC) * DIM, DIM, HC, DIM, DIM, HID, 7u, 3u);
#else
    __syncthreads();
#endif

    // GEMM2: acc(64x512) += Hc(64x64) @ W2c(64x512). B fragments pipelined one
    // tile ahead so LDS latency overlaps the WMMA pipe.
#pragma unroll
    for (int ks = 0; ks < HC / 32; ++ks) {
      int kb = ks * 32;
      const unsigned short* ap = &Hc[(tm * 16 + ln16) * WP + kb + half * 16];
      v16bf a = ld_frag16(ap, ap + 8);
      const unsigned short* bp0 = &W2s[(cg + ln16) * WP + kb + half * 16];
      v16bf bcur = ld_frag16(bp0, bp0 + 8);
#pragma unroll
      for (int tn = 0; tn < 16; ++tn) {
        v16bf bnext = bcur;
        if (tn < 15) {
          const unsigned short* bq = &W2s[(cg + (tn + 1) * 16 + ln16) * WP + kb + half * 16];
          bnext = ld_frag16(bq, bq + 8);
        }
        acc[tn] = __builtin_amdgcn_wmma_f32_16x16x32_bf16(false, a, false, bcur, (short)0, acc[tn], false, false);
        bcur = bnext;
      }
    }

#if HAVE_TDM
    __syncthreads();                                           // W2s + Hc free
    if (wv == 0 && hc + HC < HID) {
      tdm_load_2d(w2s_off, w2e + (hc + HC), HC, OUTD, HID, HID, OUTD, 4u, 3u);
      __builtin_amdgcn_s_wait_tensorcnt((short)1);             // W1(hc+HC) landed
    }
    __syncthreads();                                           // W1s ready for next GEMM1
#else
    __syncthreads();
#endif
  }

  // Epilogue: + b2, scale by gate weight, scatter-add (tokens may be in 2 experts).
#pragma unroll
  for (int tn = 0; tn < 16; ++tn) {
    int o = cg + tn * 16 + ln16;
    float bias = b2[e * OUTD + o];
#pragma unroll
    for (int v = 0; v < 8; ++v) {
      int m = tm * 16 + half * 8 + v;
      float wtv = wgt_s[m];
      if (wtv != 0.f) {
        atomicAdd(&out[(size_t)idx_s[m] * OUTD + o], (acc[tn][v] + bias) * wtv);
      }
    }
  }
}

// ---------------------------------------------------------------------------
extern "C" void kernel_launch(void* const* d_in, const int* in_sizes, int n_in,
                              void* d_out, int out_size, void* d_ws, size_t ws_size,
                              hipStream_t stream) {
  const float* x  = (const float*)d_in[0];
  const float* gw = (const float*)d_in[1];
  const float* gb = (const float*)d_in[2];
  const float* w1 = (const float*)d_in[3];
  const float* b1 = (const float*)d_in[4];
  const float* w2 = (const float*)d_in[5];
  const float* b2 = (const float*)d_in[6];

  float* out_comb  = (float*)d_out;                              // [T][O]
  float* out_probs = (float*)d_out + (size_t)TOKENS * OUTD;      // [T][E]

  char* ws = (char*)d_ws;
  size_t off = 0;
  auto take = [&](size_t bytes) { char* p = ws + off; off = (off + bytes + 15) & ~(size_t)15; return p; };
  int*            i1v  = (int*)take((size_t)TOKENS * 4);
  int*            i2v  = (int*)take((size_t)TOKENS * 4);
  float*          wt1  = (float*)take((size_t)TOKENS * 4);
  float*          wt2  = (float*)take((size_t)TOKENS * 4);
  int*            idxb = (int*)take((size_t)NEXP * CAP * 4);
  float*          wgtb = (float*)take((size_t)NEXP * CAP * 4);
  unsigned short* xbf  = (unsigned short*)take((size_t)TOKENS * DIM * 2);
  unsigned short* w1bf = (unsigned short*)take((size_t)NEXP * DIM * HID * 2);
  unsigned short* w2bf = (unsigned short*)take((size_t)NEXP * HID * OUTD * 2);

  hipMemsetAsync(out_comb, 0, (size_t)TOKENS * OUTD * sizeof(float), stream);

  cvt_bf16_kernel<<<(TOKENS * DIM) / 256, 256, 0, stream>>>(x, xbf, TOKENS * DIM);
  // w1: [E][D][H] -> [E][H][D]
  transpose_bf16_kernel<<<dim3(HID / 32, DIM / 32, NEXP), dim3(32, 8), 0, stream>>>(w1, w1bf, DIM, HID);
  // w2: [E][H][O] -> [E][O][H]
  transpose_bf16_kernel<<<dim3(OUTD / 32, HID / 32, NEXP), dim3(32, 8), 0, stream>>>(w2, w2bf, HID, OUTD);

  gate_kernel<<<TOKENS / 256, 256, 0, stream>>>(x, gw, gb, out_probs, i1v, i2v, wt1, wt2);
  dispatch_kernel<<<NEXP, 32, 0, stream>>>(i1v, i2v, wt1, wt2, idxb, wgtb);

  moe_ffn_kernel<<<dim3(CAP / 64, NEXP), 256, 0, stream>>>(xbf, w1bf, b1, w2bf, b2, idxb, wgtb, out_comb);
}
